// TConformer_61735859912775
// MI455X (gfx1250) — matrix operand order repaired and checked by reading, not compiled, not run on hardware
//
#include <hip/hip_runtime.h>
#include <math.h>

// ---------------------------------------------------------------------------
// Types for CDNA5 WMMA (wave32): v_wmma_f32_16x16x32_bf16
// ---------------------------------------------------------------------------
typedef __bf16 bf16;
typedef __attribute__((ext_vector_type(8)))  __bf16 v8bf;
typedef __attribute__((ext_vector_type(16))) __bf16 v16bf;
typedef __attribute__((ext_vector_type(8)))  float  v8f;

union Frag {
    v16bf v;
    v8bf  h[2];
};

__device__ __forceinline__ v8f wmma_bf16(v16bf a, v16bf b, v8f c) {
    // (neg_a, A, neg_b, B, c_mod, C, reuse_a, reuse_b)
    return __builtin_amdgcn_wmma_f32_16x16x32_bf16(false, a, false, b,
                                                   (short)0, c, false, false);
}

#define NDIM     256
#define NUM_HEAD 8
#define HEAD_DIM 32
#define SEQ_T    512
#define FEAT_F   64
#define BATCH_B  2
#define ATTN_W   128
#define FFN_F    1024
#define M_ROWS   (BATCH_B * FEAT_F * SEQ_T)   // 65536

// ---------------------------------------------------------------------------
// fp32 -> bf16 weight conversion
// ---------------------------------------------------------------------------
__global__ void k_cvt(const float* __restrict__ s, bf16* __restrict__ d, int n) {
    int i = blockIdx.x * 256 + threadIdx.x;
    if (i < n) d[i] = (bf16)s[i];
}

// ---------------------------------------------------------------------------
// x (B,C,T,F) -> xs (N=B*F, T, C), fp32 + bf16, LDS tiled transpose (c<->f)
// grid: (T, C/32, B), block 256
// ---------------------------------------------------------------------------
__global__ __launch_bounds__(256) void k_permute(const float* __restrict__ x,
                                                 float* __restrict__ xsF,
                                                 bf16* __restrict__ xsB) {
    const int C = NDIM, T = SEQ_T, F = FEAT_F;
    int t  = blockIdx.x;
    int c0 = blockIdx.y * 32;
    int b  = blockIdx.z;
    __shared__ float ld[32][65];
    int tid = threadIdx.x;
#pragma unroll
    for (int p = 0; p < 8; ++p) {
        int cl = p * 4 + (tid >> 6);
        int f  = tid & 63;
        ld[cl][f] = x[(((size_t)(b * C + c0 + cl)) * T + t) * F + f];
    }
    __syncthreads();
#pragma unroll
    for (int p = 0; p < 8; ++p) {
        int f  = p * 8 + (tid >> 5);
        int cl = tid & 31;
        float v = ld[cl][f];
        size_t o = ((size_t)((b * F + f) * T + t)) * C + c0 + cl;
        xsF[o] = v;
        xsB[o] = (bf16)v;
    }
}

// stage (N,T,C) fp32 -> out (B,C,T,F)
__global__ __launch_bounds__(256) void k_unpermute(const float* __restrict__ stage,
                                                   float* __restrict__ out) {
    const int C = NDIM, T = SEQ_T, F = FEAT_F;
    int t  = blockIdx.x;
    int c0 = blockIdx.y * 32;
    int b  = blockIdx.z;
    __shared__ float ld[32][65];
    int tid = threadIdx.x;
#pragma unroll
    for (int p = 0; p < 8; ++p) {
        int f  = p * 8 + (tid >> 5);
        int cl = tid & 31;
        ld[cl][f] = stage[((size_t)((b * F + f) * T + t)) * C + c0 + cl];
    }
    __syncthreads();
#pragma unroll
    for (int p = 0; p < 8; ++p) {
        int cl = p * 4 + (tid >> 6);
        int f  = tid & 63;
        out[(((size_t)(b * C + c0 + cl)) * T + t) * F + f] = ld[cl][f];
    }
}

// ---------------------------------------------------------------------------
// Generic bf16 WMMA GEMM:  out[m,n] = act( sum_k A[m,k]*W[n,k] + bias[n] )
// Block tile 128x128 (8 waves, wave tile 32x64 = 2x4 WMMA), K-step 32,
// double-buffered LDS with register-level prefetch of the next K tile.
// act: 0 = none, 1 = exact GELU.
// ---------------------------------------------------------------------------
__global__ __launch_bounds__(256) void k_gemm(const bf16* __restrict__ A,
                                              const bf16* __restrict__ W,
                                              const float* __restrict__ bias,
                                              float* __restrict__ outF,
                                              bf16* __restrict__ outB,
                                              int M, int K, int Nout, int act) {
    (void)M;
    __shared__ bf16 sA[2][128 * 40] __attribute__((aligned(16)));
    __shared__ bf16 sB[2][128 * 40] __attribute__((aligned(16)));
    int tid   = threadIdx.x;
    int lane  = tid & 31;
    int wave  = tid >> 5;
    int waveM = wave & 3;   // 4 waves over 128 rows (32 each)
    int waveN = wave >> 2;  // 2 waves over 128 cols (64 each)
    int lm = lane & 15;
    int hi = lane >> 4;
    int m0 = blockIdx.x * 128;
    int n0 = blockIdx.y * 128;

    // staging: each thread moves 16 bf16 of A and 16 bf16 of W per K-step
    int row = tid >> 1, half = tid & 1;
    const bf16* aSrc = A + (size_t)(m0 + row) * K + half * 16;
    const bf16* bSrc = W + (size_t)(n0 + row) * K + half * 16;
    int sOff = row * 40 + half * 16;

    v8bf ra0 = *(const v8bf*)(aSrc);
    v8bf ra1 = *(const v8bf*)(aSrc + 8);
    v8bf rb0 = *(const v8bf*)(bSrc);
    v8bf rb1 = *(const v8bf*)(bSrc + 8);
    *(v8bf*)&sA[0][sOff]     = ra0;
    *(v8bf*)&sA[0][sOff + 8] = ra1;
    *(v8bf*)&sB[0][sOff]     = rb0;
    *(v8bf*)&sB[0][sOff + 8] = rb1;

    v8f acc[2][4] = {};
    const int nsteps = K >> 5;
    int cur = 0;
    for (int s = 0; s < nsteps; ++s) {
        __syncthreads();
        bool more = (s + 1 < nsteps);
        if (more) {  // global->reg prefetch of next K tile, hidden behind WMMA
            const bf16* a2 = aSrc + (size_t)(s + 1) * 32;
            const bf16* b2 = bSrc + (size_t)(s + 1) * 32;
            ra0 = *(const v8bf*)(a2);
            ra1 = *(const v8bf*)(a2 + 8);
            rb0 = *(const v8bf*)(b2);
            rb1 = *(const v8bf*)(b2 + 8);
        }

        Frag fa[2], fb[4];
        const bf16* cA = sA[cur];
        const bf16* cB = sB[cur];
#pragma unroll
        for (int i = 0; i < 2; ++i) {
            // A frag: lane = M, VGPRs = K[hi*16 .. +15] contiguous
            int r0 = waveM * 32 + i * 16 + lm;
            int kh = hi * 16;
            fa[i].h[0] = *(const v8bf*)&cA[r0 * 40 + kh];
            fa[i].h[1] = *(const v8bf*)&cA[r0 * 40 + kh + 8];
        }
#pragma unroll
        for (int j = 0; j < 4; ++j) {
            // B frag: lane = N, VGPRs = K[hi*16 .. +15] contiguous
            int c0 = waveN * 64 + j * 16 + lm;
            int kh = hi * 16;
            fb[j].h[0] = *(const v8bf*)&cB[c0 * 40 + kh];
            fb[j].h[1] = *(const v8bf*)&cB[c0 * 40 + kh + 8];
        }
#pragma unroll
        for (int i = 0; i < 2; ++i)
#pragma unroll
            for (int j = 0; j < 4; ++j)
                acc[i][j] = wmma_bf16(fa[i].v, fb[j].v, acc[i][j]);

        if (more) {  // reg->LDS into the other buffer
            int nxt = cur ^ 1;
            *(v8bf*)&sA[nxt][sOff]     = ra0;
            *(v8bf*)&sA[nxt][sOff + 8] = ra1;
            *(v8bf*)&sB[nxt][sOff]     = rb0;
            *(v8bf*)&sB[nxt][sOff + 8] = rb1;
        }
        cur ^= 1;
    }

    // epilogue: C/D layout -> lane 0-15: M=r, lane 16-31: M=8+r ; N = lane&15
#pragma unroll
    for (int j = 0; j < 4; ++j) {
        int col  = n0 + waveN * 64 + j * 16 + lm;
        float bs = bias ? bias[col] : 0.0f;
#pragma unroll
        for (int i = 0; i < 2; ++i) {
#pragma unroll
            for (int r = 0; r < 8; ++r) {
                int rowO = m0 + waveM * 32 + i * 16 + r + 8 * hi;
                float v = acc[i][j][r] + bs;
                if (act == 1) v = 0.5f * v * (1.0f + erff(v * 0.70710678118654752f));
                size_t o = (size_t)rowO * Nout + col;
                if (outF) outF[o] = v;
                if (outB) outB[o] = (bf16)v;
            }
        }
    }
}

// ---------------------------------------------------------------------------
// Windowed causal flash attention. One wave per (n, h, 16-query block).
// qkv bf16 (N,T,3C) with q at +0, k at +C, v at +2C ; ctx bf16 (N,T,C).
// ---------------------------------------------------------------------------
__global__ __launch_bounds__(32) void k_attn(const bf16* __restrict__ qkv,
                                             bf16* __restrict__ ctx) {
    const int T = SEQ_T, H3C = 3 * NDIM;
    int q0   = blockIdx.x * 16;
    int h    = blockIdx.y;
    int n    = blockIdx.z;
    int lane = threadIdx.x;
    int lm = lane & 15, hi = lane >> 4;

    __shared__ bf16 pLDS[16 * 40] __attribute__((aligned(16)));
    __shared__ bf16 vLDS[32 * 40] __attribute__((aligned(16)));

    const bf16* base = qkv + (size_t)n * T * H3C;
    const int hOff = h * HEAD_DIM;

    Frag fq;  // Q as A-matrix (16 queries x Dh=32)
    {
        const bf16* qp = base + (size_t)(q0 + lm) * H3C + hOff;
        int kh = hi * 16;
        fq.h[0] = *(const v8bf*)(qp + kh);
        fq.h[1] = *(const v8bf*)(qp + kh + 8);
    }

    v8f c0v = {}, c1v = {};
    float mrow[8], lrow[8];
#pragma unroll
    for (int r = 0; r < 8; ++r) { mrow[r] = -1e30f; lrow[r] = 0.0f; }

    int kb0 = q0 - (ATTN_W - 1);
    if (kb0 < 0) kb0 = 0;
    kb0 &= ~31;
    const float sc = 0.17677669529663687f;  // 1/sqrt(Dh)

    for (int kb = kb0; kb <= q0 + 15; kb += 32) {
        // K fragments (B-matrix: lane = key, VGPRs = Dh[hi*16 .. +15])
        Frag fk0, fk1;
        {
            int kh = hi * 16;
            const bf16* kp0 = base + (size_t)(kb + lm) * H3C + NDIM + hOff + kh;
            fk0.h[0] = *(const v8bf*)(kp0);
            fk0.h[1] = *(const v8bf*)(kp0 + 8);
            const bf16* kp1 = base + (size_t)(kb + 16 + lm) * H3C + NDIM + hOff + kh;
            fk1.h[0] = *(const v8bf*)(kp1);
            fk1.h[1] = *(const v8bf*)(kp1 + 8);
        }
        v8f s0 = {}, s1 = {};
        s0 = wmma_bf16(fq.v, fk0.v, s0);
        s1 = wmma_bf16(fq.v, fk1.v, s1);

        // stage V block transposed into LDS: vLDS[d][key]
        {
            const bf16* vp = base + (size_t)(kb + lane) * H3C + 2 * NDIM + hOff;
            v8bf va = *(const v8bf*)(vp);
            v8bf vb = *(const v8bf*)(vp + 8);
            v8bf vc = *(const v8bf*)(vp + 16);
            v8bf vd = *(const v8bf*)(vp + 24);
#pragma unroll
            for (int d = 0; d < 8; ++d) {
                vLDS[(d)      * 40 + lane] = va[d];
                vLDS[(8 + d)  * 40 + lane] = vb[d];
                vLDS[(16 + d) * 40 + lane] = vc[d];
                vLDS[(24 + d) * 40 + lane] = vd[d];
            }
        }

        // online softmax (per row reductions across the 16 lanes of a half)
#pragma unroll
        for (int r = 0; r < 8; ++r) {
            int m   = q0 + r + 8 * hi;
            int ci0 = kb + lm;
            int ci1 = ci0 + 16;
            bool msk0 = (ci0 > m) || (ci0 <= m - ATTN_W);
            bool msk1 = (ci1 > m) || (ci1 <= m - ATTN_W);
            float a0 = msk0 ? -1e30f : s0[r] * sc;
            float a1 = msk1 ? -1e30f : s1[r] * sc;
            float t = fmaxf(a0, a1);
            t = fmaxf(t, __shfl_xor(t, 1));
            t = fmaxf(t, __shfl_xor(t, 2));
            t = fmaxf(t, __shfl_xor(t, 4));
            t = fmaxf(t, __shfl_xor(t, 8));
            float nm    = fmaxf(mrow[r], t);
            float scale = __expf(mrow[r] - nm);
            mrow[r] = nm;
            float p0 = msk0 ? 0.0f : __expf(a0 - nm);
            float p1 = msk1 ? 0.0f : __expf(a1 - nm);
            float rs = p0 + p1;
            rs += __shfl_xor(rs, 1);
            rs += __shfl_xor(rs, 2);
            rs += __shfl_xor(rs, 4);
            rs += __shfl_xor(rs, 8);
            lrow[r] = lrow[r] * scale + rs;
            c0v[r] *= scale;
            c1v[r] *= scale;
            int rr = r + 8 * hi;  // C-layout row -> P row
            pLDS[rr * 40 + lm]      = (bf16)p0;
            pLDS[rr * 40 + 16 + lm] = (bf16)p1;
        }

        // P (A-matrix 16x32 over keys) x V (B-matrix 32 keys x 16 dims) x2
        Frag fp, fv0, fv1;
        {
            int kh = hi * 16;
            fp.h[0] = *(const v8bf*)&pLDS[lm * 40 + kh];
            fp.h[1] = *(const v8bf*)&pLDS[lm * 40 + kh + 8];
        }
        {
            int kh = hi * 16;
            fv0.h[0] = *(const v8bf*)&vLDS[lm * 40 + kh];
            fv0.h[1] = *(const v8bf*)&vLDS[lm * 40 + kh + 8];
            fv1.h[0] = *(const v8bf*)&vLDS[(16 + lm) * 40 + kh];
            fv1.h[1] = *(const v8bf*)&vLDS[(16 + lm) * 40 + kh + 8];
        }
        c0v = wmma_bf16(fp.v, fv0.v, c0v);
        c1v = wmma_bf16(fp.v, fv1.v, c1v);
    }

#pragma unroll
    for (int r = 0; r < 8; ++r) {
        int m = q0 + r + 8 * hi;
        bf16* op = ctx + ((size_t)(n * T + m)) * NDIM + hOff;
        float inv = 1.0f / lrow[r];
        op[lm]      = (bf16)(c0v[r] * inv);
        op[16 + lm] = (bf16)(c1v[r] * inv);
    }
}

// ---------------------------------------------------------------------------
// y = LN(a + b) over C=256 ; one block (256 thr) per row.
// ---------------------------------------------------------------------------
__global__ __launch_bounds__(256) void k_add_ln(const float* __restrict__ a,
                                                const float* __restrict__ b,
                                                const float* __restrict__ g,
                                                const float* __restrict__ beta,
                                                float* __restrict__ outF,
                                                bf16* __restrict__ outB) {
    const int C = NDIM;
    size_t row = blockIdx.x;
    int tid = threadIdx.x;
    float v = a[row * C + tid] + b[row * C + tid];
    __shared__ float redS[8], redQ[8];
    float s = v, q = v * v;
#pragma unroll
    for (int msk = 1; msk < 32; msk <<= 1) {
        s += __shfl_xor(s, msk);
        q += __shfl_xor(q, msk);
    }
    if ((tid & 31) == 0) { redS[tid >> 5] = s; redQ[tid >> 5] = q; }
    __syncthreads();
    float ts = 0.0f, tq = 0.0f;
#pragma unroll
    for (int i = 0; i < 8; ++i) { ts += redS[i]; tq += redQ[i]; }
    float mu  = ts * (1.0f / C);
    float var = tq * (1.0f / C) - mu * mu;
    float y = (v - mu) * rsqrtf(var + 1e-5f) * g[tid] + beta[tid];
    outF[row * C + tid] = y;
    if (outB) outB[row * C + tid] = (bf16)y;
}

// ---------------------------------------------------------------------------
// Orchestration
// ---------------------------------------------------------------------------
extern "C" void kernel_launch(void* const* d_in, const int* in_sizes, int n_in,
                              void* d_out, int out_size, void* d_ws, size_t ws_size,
                              hipStream_t stream) {
    (void)in_sizes; (void)n_in; (void)out_size; (void)ws_size;
    const float* x          = (const float*)d_in[0];
    const float* in_proj_w  = (const float*)d_in[1];
    const float* in_proj_b  = (const float*)d_in[2];
    const float* out_proj_w = (const float*)d_in[3];
    const float* out_proj_b = (const float*)d_in[4];
    const float* ln1_g      = (const float*)d_in[5];
    const float* ln1_b      = (const float*)d_in[6];
    const float* w1         = (const float*)d_in[7];
    const float* b1         = (const float*)d_in[8];
    const float* w2         = (const float*)d_in[9];
    const float* b2         = (const float*)d_in[10];
    const float* ln2_g      = (const float*)d_in[11];
    const float* ln2_b      = (const float*)d_in[12];

    const size_t M = M_ROWS;  // 65536 rows of C=256
    unsigned char* ws = (unsigned char*)d_ws;
    size_t off = 0;
    auto carve = [&](size_t bytes) -> void* {
        void* p = ws + off;
        off += (bytes + 255) & ~(size_t)255;
        return p;
    };
    float* xsF   = (float*)carve(M * NDIM * 4);
    bf16*  xsB   = (bf16*)carve(M * NDIM * 2);
    bf16*  wInB  = (bf16*)carve((size_t)3 * NDIM * NDIM * 2);
    bf16*  wOutB = (bf16*)carve((size_t)NDIM * NDIM * 2);
    bf16*  w1B   = (bf16*)carve((size_t)FFN_F * NDIM * 2);
    bf16*  w2B   = (bf16*)carve((size_t)NDIM * FFN_F * 2);
    bf16*  qkvB  = (bf16*)carve(M * 3 * NDIM * 2);
    bf16*  ctxB  = (bf16*)carve(M * NDIM * 2);
    float* attF  = (float*)carve(M * NDIM * 4);
    float* n1F   = (float*)carve(M * NDIM * 4);
    bf16*  n1B   = (bf16*)carve(M * NDIM * 2);
    bf16*  hB    = (bf16*)carve(M * FFN_F * 2);
    float* ffwF   = attF;  // reuse: attF dead after LN1
    float* stageF = xsF;   // reuse: xs dead after LN1

    // 1) layout: x -> xs
    k_permute<<<dim3(SEQ_T, NDIM / 32, BATCH_B), 256, 0, stream>>>(x, xsF, xsB);
    // 2) weights -> bf16
    k_cvt<<<(3 * NDIM * NDIM + 255) / 256, 256, 0, stream>>>(in_proj_w, wInB, 3 * NDIM * NDIM);
    k_cvt<<<(NDIM * NDIM + 255) / 256, 256, 0, stream>>>(out_proj_w, wOutB, NDIM * NDIM);
    k_cvt<<<(FFN_F * NDIM + 255) / 256, 256, 0, stream>>>(w1, w1B, FFN_F * NDIM);
    k_cvt<<<(NDIM * FFN_F + 255) / 256, 256, 0, stream>>>(w2, w2B, NDIM * FFN_F);
    // 3) qkv = xs @ in_proj_w^T + b  (bf16 out)
    k_gemm<<<dim3(M / 128, (3 * NDIM) / 128), 256, 0, stream>>>(
        xsB, wInB, in_proj_b, nullptr, qkvB, (int)M, NDIM, 3 * NDIM, 0);
    // 4) windowed causal attention
    k_attn<<<dim3(SEQ_T / 16, NUM_HEAD, BATCH_B * FEAT_F), 32, 0, stream>>>(qkvB, ctxB);
    // 5) att_out = ctx @ out_proj_w^T + b  (fp32 out)
    k_gemm<<<dim3(M / 128, NDIM / 128), 256, 0, stream>>>(
        ctxB, wOutB, out_proj_b, attF, nullptr, (int)M, NDIM, NDIM, 0);
    // 6) norm1 = LN(xs + att_out)
    k_add_ln<<<(unsigned)M, 256, 0, stream>>>(xsF, attF, ln1_g, ln1_b, n1F, n1B);
    // 7) h = gelu(norm1 @ w1^T + b1)  (bf16 out)
    k_gemm<<<dim3(M / 128, FFN_F / 128), 256, 0, stream>>>(
        n1B, w1B, b1, nullptr, hB, (int)M, NDIM, FFN_F, 1);
    // 8) ffw = h @ w2^T + b2  (fp32 out)
    k_gemm<<<dim3(M / 128, NDIM / 128), 256, 0, stream>>>(
        hB, w2B, b2, ffwF, nullptr, (int)M, FFN_F, NDIM, 0);
    // 9) out = LN(norm1 + ffw), staged in (N,T,C)
    k_add_ln<<<(unsigned)M, 256, 0, stream>>>(n1F, ffwF, ln2_g, ln2_b, stageF, nullptr);
    // 10) stage -> (B,C,T,F)
    k_unpermute<<<dim3(SEQ_T, NDIM / 32, BATCH_B), 256, 0, stream>>>(stageF, (float*)d_out);
}
